// PointConv_52218212385431
// MI455X (gfx1250) — matrix operand order repaired and checked by reading, not compiled, not run on hardware
//
#include <hip/hip_runtime.h>

// ---------------------------------------------------------------------------
// PointConv-style pipeline for MI455X (gfx1250, wave32, WMMA).
// Shapes: B=8, C_IN=6, N=4096, M=1024, K=32.
// All WMMA operands are stored in fragment-ready layout so each lane loads its
// 16 halves as two b128 accesses (LDS or global), not 16 scalar b16 accesses.
// ---------------------------------------------------------------------------

typedef __attribute__((ext_vector_type(16))) _Float16 v16h;
typedef __attribute__((ext_vector_type(8)))  float    v8f;

#define NB   4096
#define MB   1024
#define KNN  32
#define BNS  0.9999950000374997f   // 1/sqrt(1+1e-5)

// ---- WMMA fragment k order used in this file (contiguous, not AMD's table order) ------------------
// A (16x32 f16): lane L -> row m=L&15, group g=L>>4; half i -> k:
//   k = i + 16*g
// Inverse: given k -> (g, i):
__device__ __forceinline__ void kinv_a(int k, int& g, int& i) {
  g = k >> 4;
  i = k & 15;
}
// B (32x16 f16): lane L -> col n=L&15, group g=L>>4; half i -> k = i + 16*g.

// Fragment-ready loads: lane's 16 halves contiguous (32B -> 2 x b128).
__device__ __forceinline__ v16h load_packed(const _Float16* frag, int lane) {
  return *(const v16h*)(frag + (lane << 4));
}
// B-fragment where rows-of-columns layout is [(g*ldrow + n)*16 + i]
__device__ __forceinline__ v16h load_bpk(const _Float16* frag, int ldrow, int lane) {
  const int n = lane & 15, g = lane >> 4;
  return *(const v16h*)(frag + ((g * ldrow + n) << 4));
}
__device__ __forceinline__ v8f wmma_f16(v16h a, v16h b, v8f c) {
  return __builtin_amdgcn_wmma_f32_16x16x32_f16(false, a, false, b, (short)0, c,
                                                false, false);
}

// ---------------------------------------------------------------------------
// K1: KDE density + fused 1->8->8->1 sigmoid MLP -> s[B,N]
// grid = B * (N/256), block = 256
// ---------------------------------------------------------------------------
__global__ void kde_scale_kernel(const float* __restrict__ x,
                                 const float* sw1, const float* sb1,
                                 const float* sg1, const float* sbt1,
                                 const float* sw2, const float* sb2,
                                 const float* sg2, const float* sbt2,
                                 const float* sw3, const float* sb3,
                                 float* __restrict__ s_out) {
  const int b = blockIdx.x >> 4;
  const int n = (blockIdx.x & 15) * 256 + threadIdx.x;
  __shared__ float tx[256], ty[256], tz[256];
  const float* xb = x + (size_t)b * 6 * NB;
  const float px = xb[n], py = xb[NB + n], pz = xb[2 * NB + n];
  float acc = 0.0f;
  for (int t = 0; t < 16; ++t) {
    const int j = t * 256 + threadIdx.x;
    tx[threadIdx.x] = xb[j];
    ty[threadIdx.x] = xb[NB + j];
    tz[threadIdx.x] = xb[2 * NB + j];
    __syncthreads();
#pragma unroll 4
    for (int i = 0; i < 256; ++i) {
      const float dx = px - tx[i], dy = py - ty[i], dz = pz - tz[i];
      acc += __expf(-0.5f * (dx * dx + dy * dy + dz * dz));
    }
    __syncthreads();
  }
  const float dens = acc * (1.0f / (float)NB) * 0.06349363593424098f; // /(2pi)^1.5
  const float si = 1.0f / dens;
  float a1[8], a2[8];
#pragma unroll
  for (int j = 0; j < 8; ++j) {
    float v = sw1[j] * si + sb1[j];
    v = v * (sg1[j] * BNS) + sbt1[j];
    a1[j] = fmaxf(v, 0.0f);
  }
#pragma unroll
  for (int i = 0; i < 8; ++i) {
    float v = sb2[i];
#pragma unroll
    for (int j = 0; j < 8; ++j) v += sw2[i * 8 + j] * a1[j];
    v = v * (sg2[i] * BNS) + sbt2[i];
    a2[i] = fmaxf(v, 0.0f);
  }
  float o = sb3[0];
#pragma unroll
  for (int j = 0; j < 8; ++j) o += sw3[j] * a2[j];
  s_out[(size_t)b * NB + n] = 1.0f / (1.0f + __expf(-o));
}

// ---------------------------------------------------------------------------
// K2: farthest point sampling. grid = B, block = 1024, dynamic LDS.
// Also writes q[B,M,3] and output channels 0..2 (qc).
// ---------------------------------------------------------------------------
__global__ void fps_kernel(const float* __restrict__ x, int* __restrict__ fpsidx,
                           float* __restrict__ q, float* __restrict__ out) {
  extern __shared__ char smem[];
  float* px   = (float*)smem;          // 4096
  float* py   = px + NB;               // 4096
  float* pz   = py + NB;               // 4096
  float* mind = pz + NB;               // 4096
  float* rv   = mind + NB;             // 1024
  int*   ri   = (int*)(rv + 1024);     // 1024
  int*   sel  = ri + 1024;             // 1024

  const int b = blockIdx.x, tid = threadIdx.x;
  const float* xb = x + (size_t)b * 6 * NB;
  for (int j = tid; j < NB; j += 1024) {
    px[j] = xb[j]; py[j] = xb[NB + j]; pz[j] = xb[2 * NB + j];
    mind[j] = 1e10f;
  }
  if (tid == 0) sel[0] = 0;
  __syncthreads();

  int last = 0;
  for (int it = 1; it < MB; ++it) {
    const float lx = px[last], ly = py[last], lz = pz[last];
    float bestv = -1.0f; int besti = 0;
    for (int j = tid; j < NB; j += 1024) {
      const float dx = px[j] - lx, dy = py[j] - ly, dz = pz[j] - lz;
      const float d = dx * dx + dy * dy + dz * dz;
      const float m = fminf(mind[j], d);
      mind[j] = m;
      if (m > bestv) { bestv = m; besti = j; }
    }
    rv[tid] = bestv; ri[tid] = besti;
    __syncthreads();
    for (int s = 512; s > 0; s >>= 1) {
      if (tid < s) {
        const float ov = rv[tid + s]; const int oi = ri[tid + s];
        if (ov > rv[tid] || (ov == rv[tid] && oi < ri[tid])) { rv[tid] = ov; ri[tid] = oi; }
      }
      __syncthreads();
    }
    last = ri[0];                       // argmax (first max on ties)
    if (tid == 0) sel[it] = last;
    __syncthreads();
  }

  const int id = sel[tid];
  fpsidx[b * MB + tid] = id;
  const float qx = px[id], qy = py[id], qz = pz[id];
  float* qp = q + ((size_t)b * MB + tid) * 3;
  qp[0] = qx; qp[1] = qy; qp[2] = qz;
  float* ob = out + (size_t)b * 131 * MB;
  ob[0 * MB + tid] = qx; ob[1 * MB + tid] = qy; ob[2 * MB + tid] = qz;
}

// ---------------------------------------------------------------------------
// K3: 32-NN per query (order-invariant downstream). grid = B*M/128, block=128.
// ---------------------------------------------------------------------------
__global__ void knn_kernel(const float* __restrict__ x, const float* __restrict__ q,
                           int* __restrict__ knn) {
  const int g = blockIdx.x * 128 + threadIdx.x;
  const int b = g >> 10, m = g & 1023;
  const float* xb = x + (size_t)b * 6 * NB;
  const float* qp = q + ((size_t)b * MB + m) * 3;
  const float qx = qp[0], qy = qp[1], qz = qp[2];
  float bd[KNN]; int bi[KNN];
#pragma unroll
  for (int k = 0; k < KNN; ++k) { bd[k] = 3.4e38f; bi[k] = 0; }
  for (int n = 0; n < NB; ++n) {
    const float dx = xb[n] - qx, dy = xb[NB + n] - qy, dz = xb[2 * NB + n] - qz;
    const float d = dx * dx + dy * dy + dz * dz;
    if (d < bd[KNN - 1]) {
      int kpos = KNN - 1;
      while (kpos > 0 && bd[kpos - 1] > d) {
        bd[kpos] = bd[kpos - 1]; bi[kpos] = bi[kpos - 1]; --kpos;
      }
      bd[kpos] = d; bi[kpos] = n;
    }
  }
  int* kb = knn + ((size_t)b * MB + m) * KNN;
#pragma unroll
  for (int k = 0; k < KNN; ++k) kb[k] = bi[k];
}

// ---------------------------------------------------------------------------
// K4: f32 -> f16 conversion of lw and mw2, written directly in WMMA
// A-fragment-ready order. grid = 512, block = 256.
//   lwA index:  ((rt*32 + kt)*32 + row + 16*g)*16 + i   (rt=o>>4, kt=f>>5)
//   mwA index:  ((rt*2  + kt)*32 + row + 16*g)*16 + i
// ---------------------------------------------------------------------------
__global__ void convert_swizzle_kernel(const float* __restrict__ lw,
                                       const float* __restrict__ mw2,
                                       _Float16* __restrict__ lwA,
                                       _Float16* __restrict__ mwA) {
  const int e = blockIdx.x * 256 + threadIdx.x;
  if (e < 131072) {
    const int o = e >> 10, f = e & 1023;
    const int rt = o >> 4, row = o & 15;
    const int kt = f >> 5, k = f & 31;
    int g, i; kinv_a(k, g, i);
    lwA[(((rt * 32 + kt) * 32) + row + 16 * g) * 16 + i] = (_Float16)lw[e];
  }
  if (e < 4096) {
    const int o = e >> 6, c = e & 63;
    const int rt = o >> 4, row = o & 15;
    const int kt = c >> 5, k = c & 31;
    int g, i; kinv_a(k, g, i);
    mwA[(((rt * 2 + kt) * 32) + row + 16 * g) * 16 + i] = (_Float16)mw2[e];
  }
}

// ---------------------------------------------------------------------------
// K5: fused gather + w-MLP + h-MLP + agg + final projection.
// grid = B * (M/16), block = 256 (8 wave32), dynamic LDS ~197KB.
// ---------------------------------------------------------------------------
struct MainArgs {
  const float *x, *s, *q; const int* knn;
  const float *ww1, *wb1, *wg1, *wbt1, *ww2, *wb2, *wg2, *wbt2, *ww3, *wb3;
  const float *mw1, *mb1, *mg1, *mbt1, *mb2, *mg2, *mbt2;
  const float *lb, *lg, *lbt;
  const _Float16 *lwA, *mwA;
  float* out;
};

enum {
  OWW1 = 0, OWB1 = 24, OWS1 = 32, OWBT1 = 40,
  OWW2 = 48, OWB2 = 112, OWS2 = 120, OWBT2 = 128,
  OWW3 = 136, OWB3 = 264,
  OMW1 = 280, OMB1 = 664, OMS1 = 728, OMBT1 = 792,
  OMB2 = 856, OMS2 = 920, OMBT2 = 984,
  OLB = 1048, OLS = 1176, OLBT = 1304,
  WBUF_FLOATS = 1536
};

__global__ void pointconv_main_kernel(MainArgs a) {
  extern __shared__ char smem[];
  float*    wbuf = (float*)smem;                  // 1536 f
  float*    qloc = wbuf + WBUF_FLOATS;            // 64 f
  int*      nidx = (int*)(qloc + 64);             // 512 i   (8448 B so far, 32B-aligned)
  // B-fragment-ready: H1p[(cblk*512 + p)*16 + i], cblk = c>>4, i = c&15
  _Float16* H1p  = (_Float16*)(nidx + 512);       // 32768 halves
  // B-fragment-ready: Wlp[m*512 + (g*16 + w)*16 + i], k = i + 16*g
  _Float16* Wlp  = H1p + 32768;                   // 8192 halves
  // A-fragment-ready: H2p[((m*4 + ctile)*32 + lane)*16 + i]
  _Float16* H2p  = Wlp + 8192;                    // 32768 halves
  // B-fragment-ready: Gp[c*256 + m*16 + w]  (f = c*16+w; reader k = i+16g)
  _Float16* Gp   = H2p + 32768;                   // 16384 halves
  // A-fragment-ready copy of mw2 (already swizzled in global)
  _Float16* MW   = Gp + 16384;                    // 4096 halves

  const int b = blockIdx.x >> 6;
  const int m0 = (blockIdx.x & 63) * 16;
  const int tid = threadIdx.x;
  const int lane = tid & 31, wave = tid >> 5;

  // warm up L2 for the big swizzled weight matrix (global_prefetch_b8)
  __builtin_prefetch(a.lwA + (size_t)(wave * 16 + lane) * 1024, 0, 1);

  // ---- stage 0: stage weights (BN scales pre-fused), q, knn idx, mw2 ------
  for (int i = tid; i < 24; i += 256) wbuf[OWW1 + i] = a.ww1[i];
  for (int i = tid; i < 8; i += 256) {
    wbuf[OWB1 + i] = a.wb1[i];  wbuf[OWS1 + i] = a.wg1[i] * BNS;
    wbuf[OWBT1 + i] = a.wbt1[i];
    wbuf[OWB2 + i] = a.wb2[i];  wbuf[OWS2 + i] = a.wg2[i] * BNS;
    wbuf[OWBT2 + i] = a.wbt2[i];
  }
  for (int i = tid; i < 64; i += 256) {
    wbuf[OWW2 + i] = a.ww2[i];
    wbuf[OMB1 + i] = a.mb1[i];  wbuf[OMS1 + i] = a.mg1[i] * BNS;
    wbuf[OMBT1 + i] = a.mbt1[i];
    wbuf[OMB2 + i] = a.mb2[i];  wbuf[OMS2 + i] = a.mg2[i] * BNS;
    wbuf[OMBT2 + i] = a.mbt2[i];
  }
  for (int i = tid; i < 128; i += 256) {
    wbuf[OWW3 + i] = a.ww3[i];
    wbuf[OLB + i] = a.lb[i];    wbuf[OLS + i] = a.lg[i] * BNS;
    wbuf[OLBT + i] = a.lbt[i];
  }
  for (int i = tid; i < 16; i += 256) wbuf[OWB3 + i] = a.wb3[i];
  for (int i = tid; i < 384; i += 256) wbuf[OMW1 + i] = a.mw1[i];
  for (int i = tid; i < 256; i += 256)                    // 4096 halves as v16h
    ((v16h*)MW)[i] = ((const v16h*)a.mwA)[i];
  for (int i = tid; i < 48; i += 256) qloc[i] = a.q[((size_t)b * MB + m0) * 3 + i];
  for (int i = tid; i < 512; i += 256) nidx[i] = a.knn[((size_t)b * MB + m0) * KNN + i];
  __syncthreads();

  // ---- stage 1: gather + w-MLP (3->8->8->16) + h layer1 (6->64) -----------
  const float* xb = a.x + (size_t)b * 6 * NB;
  const float* sb = a.s + (size_t)b * NB;
  for (int p = tid; p < 512; p += 256) {
    const int m_l = p >> 5, k = p & 31;
    const int n = nidx[p];
    const float x0 = xb[n],          x1 = xb[NB + n],     x2 = xb[2 * NB + n];
    const float x3 = xb[3 * NB + n], x4 = xb[4 * NB + n], x5 = xb[5 * NB + n];
    const float sv = sb[n];
    const float r0 = x0 - qloc[m_l * 3 + 0];
    const float r1 = x1 - qloc[m_l * 3 + 1];
    const float r2 = x2 - qloc[m_l * 3 + 2];
    // w-MLP
    float wa[8], wc[8];
#pragma unroll
    for (int o = 0; o < 8; ++o) {
      float v = wbuf[OWB1 + o] + wbuf[OWW1 + o * 3] * r0 +
                wbuf[OWW1 + o * 3 + 1] * r1 + wbuf[OWW1 + o * 3 + 2] * r2;
      v = v * wbuf[OWS1 + o] + wbuf[OWBT1 + o];
      wa[o] = fmaxf(v, 0.0f);
    }
#pragma unroll
    for (int o = 0; o < 8; ++o) {
      float v = wbuf[OWB2 + o];
#pragma unroll
      for (int j = 0; j < 8; ++j) v += wbuf[OWW2 + o * 8 + j] * wa[j];
      v = v * wbuf[OWS2 + o] + wbuf[OWBT2 + o];
      wc[o] = fmaxf(v, 0.0f);
    }
    // last w layer (no bn/relu) -> B-fragment-ready Wlp
    {
      const int gk = k >> 4, ik = k & 15;
      _Float16* wr = Wlp + m_l * 512 + ik;
#pragma unroll
      for (int o = 0; o < 16; ++o) {
        float v = wbuf[OWB3 + o];
#pragma unroll
        for (int j = 0; j < 8; ++j) v += wbuf[OWW3 + o * 8 + j] * wc[j];
        wr[(gk * 16 + o) * 16] = (_Float16)v;
      }
    }
    // h layer1: input = concat(rel, x[3:6]) * s -> B-fragment-ready H1p
    const float h0 = r0 * sv, h1v = r1 * sv, h2v = r2 * sv;
    const float h3 = x3 * sv, h4 = x4 * sv, h5 = x5 * sv;
#pragma unroll
    for (int ob = 0; ob < 4; ++ob) {
      v16h buf;
#pragma unroll
      for (int oi = 0; oi < 16; ++oi) {
        const int o = ob * 16 + oi;
        const float* wr = wbuf + OMW1 + o * 6;
        float v = wbuf[OMB1 + o] + wr[0] * h0 + wr[1] * h1v + wr[2] * h2v +
                  wr[3] * h3 + wr[4] * h4 + wr[5] * h5;
        v = fmaxf(v * wbuf[OMS1 + o] + wbuf[OMBT1 + o], 0.0f);
        buf[oi] = (_Float16)v;
      }
      *(v16h*)(H1p + (ob * 512 + p) * 16) = buf;   // one 32B LDS store
    }
  }
  __syncthreads();

  // ---- stage 2: h layer2 GEMM  H2 = relu(bn(mw2 @ H1)),  64x512, K=64 -----
  // 4 row-tiles x 32 col-tiles = 128 tiles; wave handles 16; 2 WMMAs each.
  for (int j = 0; j < 16; ++j) {
    const int t = wave * 16 + j;
    const int rt = t & 3, ct = t >> 2;
    v8f acc = {};
#pragma unroll
    for (int kt = 0; kt < 2; ++kt) {
      const v16h A  = load_packed(MW + (rt * 2 + kt) * 512, lane);
      const v16h Bf = load_bpk(H1p + ((kt * 2) * 512 + ct * 16) * 16, 512, lane);
      acc = wmma_f16(A, Bf, acc);
    }
    const int nn = lane & 15, g = lane >> 4;
    const int pcol = ct * 16 + nn, m_l = pcol >> 5, kk = pcol & 31;
    int gr, ir; kinv_a(kk, gr, ir);
    _Float16* dst = H2p + ((m_l * 4 + rt) * 32 + 16 * gr) * 16 + ir;
#pragma unroll
    for (int r = 0; r < 8; ++r) {
      const int c = rt * 16 + r + 8 * g;
      float v = acc[r] + wbuf[OMB2 + c];
      v = fmaxf(v * wbuf[OMS2 + c] + wbuf[OMBT2 + c], 0.0f);
      dst[(c & 15) * 16] = (_Float16)v;            // A-fragment-ready scatter
    }
  }
  __syncthreads();

  // ---- stage 3: agg per query: G[c,w] = sum_k H2[c,k] * Wl[k,w]  (K=32) ---
  // 16 m x 4 c-tiles = 64 tiles; wave handles 8; 1 WMMA each.
  for (int j = 0; j < 8; ++j) {
    const int t = wave * 8 + j;
    const int m_l = t >> 2, ctile = t & 3;
    const v16h A  = load_packed(H2p + (m_l * 4 + ctile) * 512, lane);
    const v16h Bf = load_bpk(Wlp + m_l * 512, 16, lane);
    v8f acc = {};
    acc = wmma_f16(A, Bf, acc);
    const int wcol = lane & 15, g = lane >> 4;
#pragma unroll
    for (int r = 0; r < 8; ++r) {
      const int c = ctile * 16 + r + 8 * g;        // f = c*16 + wcol
      Gp[c * 256 + m_l * 16 + wcol] = (_Float16)acc[r];
    }
  }
  __syncthreads();

  // ---- stage 4: final projection: out[128,16] = L[128,1024] @ G[1024,16] --
  // 8 row-tiles, one per wave, K = 1024 -> 32 WMMAs per wave.
  {
    const int rt = wave;
    v8f acc = {};
    for (int kt = 0; kt < 32; ++kt) {
      const v16h A  = load_packed(a.lwA + (size_t)((rt * 32 + kt) * 32) * 16, lane);
      const v16h Bf = load_bpk(Gp + kt * 512, 16, lane);
      acc = wmma_f16(A, Bf, acc);
    }
    const int mcol = lane & 15, g = lane >> 4;
    float* ob = a.out + (size_t)b * 131 * MB + m0 + mcol;
#pragma unroll
    for (int r = 0; r < 8; ++r) {
      const int o = rt * 16 + r + 8 * g;
      float v = acc[r] + wbuf[OLB + o];
      v = fmaxf(v * wbuf[OLS + o] + wbuf[OLBT + o], 0.0f);
      ob[(size_t)(3 + o) * MB] = v;
    }
  }
}

// ---------------------------------------------------------------------------
extern "C" void kernel_launch(void* const* d_in, const int* in_sizes, int n_in,
                              void* d_out, int out_size, void* d_ws, size_t ws_size,
                              hipStream_t stream) {
  (void)in_sizes; (void)n_in; (void)out_size; (void)ws_size;
  const float* x    = (const float*)d_in[0];
  const float* sw1  = (const float*)d_in[1];
  const float* sb1  = (const float*)d_in[2];
  const float* sg1  = (const float*)d_in[3];
  const float* sbt1 = (const float*)d_in[4];
  const float* sw2  = (const float*)d_in[5];
  const float* sb2  = (const float*)d_in[6];
  const float* sg2  = (const float*)d_in[7];
  const float* sbt2 = (const float*)d_in[8];
  const float* sw3  = (const float*)d_in[9];
  const float* sb3  = (const float*)d_in[10];
  const float* ww1  = (const float*)d_in[11];
  const float* wb1  = (const float*)d_in[12];
  const float* wg1  = (const float*)d_in[13];
  const float* wbt1 = (const float*)d_in[14];
  const float* ww2  = (const float*)d_in[15];
  const float* wb2  = (const float*)d_in[16];
  const float* wg2  = (const float*)d_in[17];
  const float* wbt2 = (const float*)d_in[18];
  const float* ww3  = (const float*)d_in[19];
  const float* wb3  = (const float*)d_in[20];
  const float* mw1  = (const float*)d_in[21];
  const float* mb1  = (const float*)d_in[22];
  const float* mg1  = (const float*)d_in[23];
  const float* mbt1 = (const float*)d_in[24];
  const float* mw2  = (const float*)d_in[25];
  const float* mb2  = (const float*)d_in[26];
  const float* mg2  = (const float*)d_in[27];
  const float* mbt2 = (const float*)d_in[28];
  const float* lw   = (const float*)d_in[29];
  const float* lb   = (const float*)d_in[30];
  const float* lg   = (const float*)d_in[31];
  const float* lbt  = (const float*)d_in[32];
  float* out = (float*)d_out;

  // workspace layout (bytes)
  char* w = (char*)d_ws;
  float*    s_buf = (float*)w;                                       // 8*4096
  float*    q_buf = (float*)(w + (size_t)32768 * 4);                 // 8*1024*3
  int*      fps_b = (int*)(w + (size_t)(32768 + 24576) * 4);         // 8*1024
  int*      knn_b = (int*)(w + (size_t)(32768 + 24576 + 8192) * 4);  // 8*1024*32
  _Float16* lwA   = (_Float16*)(w + (size_t)(32768 + 24576 + 8192 + 262144) * 4);
  _Float16* mwA   = lwA + 131072;

  kde_scale_kernel<<<dim3(128), dim3(256), 0, stream>>>(
      x, sw1, sb1, sg1, sbt1, sw2, sb2, sg2, sbt2, sw3, sb3, s_buf);

  const size_t fps_lds = (size_t)(4 * 4096 + 1024) * 4 + (size_t)(1024 + 1024) * 4;
  fps_kernel<<<dim3(8), dim3(1024), fps_lds, stream>>>(x, fps_b, q_buf, out);

  knn_kernel<<<dim3(64), dim3(128), 0, stream>>>(x, q_buf, knn_b);

  convert_swizzle_kernel<<<dim3(512), dim3(256), 0, stream>>>(lw, mw2, lwA, mwA);

  MainArgs ma;
  ma.x = x; ma.s = s_buf; ma.q = q_buf; ma.knn = knn_b;
  ma.ww1 = ww1; ma.wb1 = wb1; ma.wg1 = wg1; ma.wbt1 = wbt1;
  ma.ww2 = ww2; ma.wb2 = wb2; ma.wg2 = wg2; ma.wbt2 = wbt2;
  ma.ww3 = ww3; ma.wb3 = wb3;
  ma.mw1 = mw1; ma.mb1 = mb1; ma.mg1 = mg1; ma.mbt1 = mbt1;
  ma.mb2 = mb2; ma.mg2 = mg2; ma.mbt2 = mbt2;
  ma.lb = lb; ma.lg = lg; ma.lbt = lbt;
  ma.lwA = lwA; ma.mwA = mwA; ma.out = out;

  const size_t main_lds =
      (size_t)(WBUF_FLOATS + 64) * 4 + (size_t)512 * 4 +
      (size_t)(32768 + 8192 + 32768 + 16384 + 4096) * 2;
  pointconv_main_kernel<<<dim3(512), dim3(256), main_lds, stream>>>(ma);
}
